// HeteroGraph_58540404244651
// MI455X (gfx1250) — matrix operand order repaired and checked by reading, not compiled, not run on hardware
//
#include <hip/hip_runtime.h>
#include <hip/hip_bf16.h>
#include <stdint.h>

#define H 128
#define NTOT 500000
#define NB 2048
#define NREL 14
#define NTYPE 7

typedef __bf16          v16bf __attribute__((ext_vector_type(16)));
typedef float           v8f   __attribute__((ext_vector_type(8)));
typedef unsigned short  v16us __attribute__((ext_vector_type(16)));
typedef unsigned short  v8us  __attribute__((ext_vector_type(8)));
typedef int             v4i   __attribute__((ext_vector_type(4)));

union BFrag { v16us u; v16bf b; v8us h[2]; };

#if __has_builtin(__builtin_amdgcn_global_load_async_to_lds_b128) && \
    __has_builtin(__builtin_amdgcn_s_wait_asynccnt)
#define ASYNC_OK 1
#endif

// 16B gather: global -> LDS.  Async (ASYNCcnt-tracked, no VGPR round trip) on
// gfx1250 toolchains that expose the builtin; otherwise b128 load + ds_store.
__device__ __forceinline__ void async_ld16(const void* g, void* l) {
#ifdef ASYNC_OK
    __builtin_amdgcn_global_load_async_to_lds_b128((v4i*)g, (v4i*)l, 0, 0);
#else
    *(uint4*)l = *(const uint4*)g;
#endif
}
__device__ __forceinline__ void async_wait() {
#ifdef ASYNC_OK
    __builtin_amdgcn_s_wait_asynccnt(0);
#endif
}

__device__ __forceinline__ unsigned short f2bf(float f) {
    unsigned int u = __float_as_uint(f);
    unsigned int r = u + 0x7FFFu + ((u >> 16) & 1u);   // round-to-nearest-even
    return (unsigned short)(r >> 16);
}

// ---------------------------------------------------------------------------
// Pack 30 weight matrices (c1_root, c1_rel[14], c2_root, c2_rel[14]) from
// row-major f32 [K=128][N=128] into bf16 WMMA B-fragment order:
//   Wpk[ ((nt*4+kt)*32 + lane)*16 + e ] = W[32*kt + (e<8 ? 8*(lane>>4)+e : 16+8*(lane>>4)+e-8)][16*nt + (lane&15)]
// so each B fragment in the GEMM kernels is one contiguous 32B load.
// ---------------------------------------------------------------------------
__global__ void pack_weights(const float* __restrict__ c1_root,
                             const float* __restrict__ c1_rel,
                             const float* __restrict__ c2_root,
                             const float* __restrict__ c2_rel,
                             unsigned short* __restrict__ Wpk) {
    int idx = blockIdx.x * blockDim.x + threadIdx.x;
    const int TOTAL = 30 * 16384;
    if (idx >= TOTAL) return;
    int mat = idx >> 14;
    int rem = idx & 16383;
    int e    =  rem        & 15;
    int lane = (rem >> 4)  & 31;
    int kt   = (rem >> 9)  & 3;
    int nt   = (rem >> 11) & 7;
    const float* W;
    if      (mat == 0)  W = c1_root;
    else if (mat <= 14) W = c1_rel + (size_t)(mat - 1) * 16384;
    else if (mat == 15) W = c2_root;
    else                W = c2_rel + (size_t)(mat - 16) * 16384;
    int k = 32 * kt + (e < 8 ? 8 * (lane >> 4) + e : 16 + 8 * (lane >> 4) + (e - 8));
    int n = 16 * nt + (lane & 15);
    Wpk[idx] = f2bf(W[k * H + n]);
}

__global__ void zero_f32(float* __restrict__ p, long long n) {
    long long i = (long long)blockIdx.x * blockDim.x + threadIdx.x;
    if (i < n) p[i] = 0.0f;
}

__global__ void count_edges(const int* __restrict__ dst, int E, int dst_off,
                            float* __restrict__ cbuf) {
    int i = blockIdx.x * blockDim.x + threadIdx.x;
    if (i < E) atomicAdd(&cbuf[dst[i] + dst_off], 1.0f);
}

__global__ void invert_counts(float* __restrict__ p, long long n) {
    long long i = (long long)blockIdx.x * blockDim.x + threadIdx.x;
    if (i < n) p[i] = 1.0f / fmaxf(p[i], 1.0f);
}

// Input projection: x[n,feat] @ W[feat,128] + b  -> bf16 features (tiny K)
__global__ void proj_type(const float* __restrict__ x, const float* __restrict__ W,
                          const float* __restrict__ bias, unsigned short* __restrict__ Xb,
                          int n, int feat, int row_off) {
    long long idx = (long long)blockIdx.x * blockDim.x + threadIdx.x;
    if (idx >= (long long)n * H) return;
    int node = (int)(idx >> 7), col = (int)(idx & 127);
    float s = bias[col];
    const float* xr = x + (size_t)node * feat;
    for (int f = 0; f < feat; ++f) s += xr[f] * W[f * H + col];
    Xb[(size_t)(row_off + node) * H + col] = f2bf(s);
}

// ---------------------------------------------------------------------------
// Root GEMM: OUT[32 rows,128] = Xb[32 rows,128] @ W + bias.  One wave / block,
// two 16-row A tiles sharing every B fragment.  NTOT % 32 == 0 -> branch-free.
// ---------------------------------------------------------------------------
__global__ void __launch_bounds__(32)
gemm_root(const unsigned short* __restrict__ Xb, const unsigned short* __restrict__ Wm,
          const float* __restrict__ bias, float* __restrict__ OUT) {
    __shared__ __attribute__((aligned(16))) unsigned short As[2][16][H];
    int lane = threadIdx.x;
    size_t row0 = (size_t)blockIdx.x * 32;
    // 8KB contiguous gather: 16 x (512B per instruction across the wave)
    {
        const unsigned short* gp = Xb + row0 * H + lane * 8;
        unsigned short*       lp = &As[0][0][0] + lane * 8;
        #pragma unroll
        for (int i = 0; i < 16; ++i) async_ld16(gp + i * 256, lp + i * 256);
    }
    async_wait();
    __syncthreads();
    int m = lane & 15, hh = lane >> 4;
    BFrag A0[4], A1[4];
    #pragma unroll
    for (int kt = 0; kt < 4; ++kt) {
        A0[kt].h[0] = *(const v8us*)&As[0][m][32 * kt + 8 * hh];
        A0[kt].h[1] = *(const v8us*)&As[0][m][32 * kt + 16 + 8 * hh];
        A1[kt].h[0] = *(const v8us*)&As[1][m][32 * kt + 8 * hh];
        A1[kt].h[1] = *(const v8us*)&As[1][m][32 * kt + 16 + 8 * hh];
    }
    int n = lane & 15, rbase = hh * 8;
    #pragma unroll
    for (int nt = 0; nt < 8; ++nt) {
        v8f acc0 = {}, acc1 = {};
        #pragma unroll
        for (int kt = 0; kt < 4; ++kt) {
            BFrag B;
            B.u = *(const v16us*)(Wm + (((nt * 4 + kt) * 32 + lane) << 4));
            acc0 = __builtin_amdgcn_wmma_f32_16x16x32_bf16(
                     false, A0[kt].b, false, B.b, (short)0, acc0, false, false);
            acc1 = __builtin_amdgcn_wmma_f32_16x16x32_bf16(
                     false, A1[kt].b, false, B.b, (short)0, acc1, false, false);
        }
        int col = nt * 16 + n;
        float bc = bias[col];
        #pragma unroll
        for (int i = 0; i < 8; ++i)
            OUT[(row0 + rbase + i) * H + col] = acc0[i] + bc;
        #pragma unroll
        for (int i = 0; i < 8; ++i)
            OUT[(row0 + 16 + rbase + i) * H + col] = acc1[i] + bc;
    }
}

// ---------------------------------------------------------------------------
// Edge GEMM for one relation: for 32 edges, OUT[dst] += inv_cnt[dst] *
// (Xb[src] @ W_rel).  Branch-free scatter: tail edges target dummy row NTOT
// with scale 0.  One wave / block; f32 atomic scatter (64B-coalesced groups).
// ---------------------------------------------------------------------------
__global__ void __launch_bounds__(32)
gemm_edges(const unsigned short* __restrict__ Xb,
           const int* __restrict__ src, const int* __restrict__ dst,
           int E, int src_off, int dst_off,
           const unsigned short* __restrict__ Wm,
           const float* __restrict__ invc, float* __restrict__ OUT) {
    __shared__ __attribute__((aligned(16))) unsigned short As[2][16][H];
    __shared__ int   dIdx[32];
    __shared__ float dScl[32];
    int lane = threadIdx.x;
    int e0 = blockIdx.x * 32;
    {   // per-lane edge metadata (dummy row for tail)
        int e = e0 + lane;
        if (e < E) { int d = dst[e] + dst_off; dIdx[lane] = d; dScl[lane] = invc[d]; }
        else       { dIdx[lane] = NTOT;        dScl[lane] = 0.0f; }
    }
    {   // async gather of 32 source rows (two 16-row tiles), 2 lanes per row
        int r = lane >> 1, half = lane & 1;
        #pragma unroll
        for (int t = 0; t < 2; ++t) {
            int e = e0 + t * 16 + r; if (e >= E) e = E - 1;
            int s = src[e] + src_off;
            const unsigned short* gp = Xb + (size_t)s * H + half * 64;
            unsigned short*       lp = &As[t][r][half * 64];
            #pragma unroll
            for (int i = 0; i < 8; ++i) async_ld16(gp + i * 8, lp + i * 8);
        }
    }
    async_wait();
    __syncthreads();
    int m = lane & 15, hh = lane >> 4;
    BFrag A0[4], A1[4];
    #pragma unroll
    for (int kt = 0; kt < 4; ++kt) {
        A0[kt].h[0] = *(const v8us*)&As[0][m][32 * kt + 8 * hh];
        A0[kt].h[1] = *(const v8us*)&As[0][m][32 * kt + 16 + 8 * hh];
        A1[kt].h[0] = *(const v8us*)&As[1][m][32 * kt + 8 * hh];
        A1[kt].h[1] = *(const v8us*)&As[1][m][32 * kt + 16 + 8 * hh];
    }
    int n = lane & 15, rbase = hh * 8;
    int dreg0[8], dreg1[8]; float sreg0[8], sreg1[8];
    #pragma unroll
    for (int i = 0; i < 8; ++i) {
        dreg0[i] = dIdx[rbase + i];      sreg0[i] = dScl[rbase + i];
        dreg1[i] = dIdx[16 + rbase + i]; sreg1[i] = dScl[16 + rbase + i];
    }
    #pragma unroll
    for (int nt = 0; nt < 8; ++nt) {
        v8f acc0 = {}, acc1 = {};
        #pragma unroll
        for (int kt = 0; kt < 4; ++kt) {
            BFrag B;
            B.u = *(const v16us*)(Wm + (((nt * 4 + kt) * 32 + lane) << 4));
            acc0 = __builtin_amdgcn_wmma_f32_16x16x32_bf16(
                     false, A0[kt].b, false, B.b, (short)0, acc0, false, false);
            acc1 = __builtin_amdgcn_wmma_f32_16x16x32_bf16(
                     false, A1[kt].b, false, B.b, (short)0, acc1, false, false);
        }
        int col = nt * 16 + n;
        #pragma unroll
        for (int i = 0; i < 8; ++i)
            atomicAdd(&OUT[(size_t)dreg0[i] * H + col], acc0[i] * sreg0[i]);
        #pragma unroll
        for (int i = 0; i < 8; ++i)
            atomicAdd(&OUT[(size_t)dreg1[i] * H + col], acc1[i] * sreg1[i]);
    }
}

// ELU + LayerNorm over H=128; one wave per row (4 cols/lane), wave32 shuffles.
__global__ void act_ln(const float* __restrict__ IN, const float* __restrict__ g,
                       const float* __restrict__ b, float* __restrict__ XA,
                       unsigned short* __restrict__ Xb, int nrows) {
    int wave = threadIdx.x >> 5, lane = threadIdx.x & 31;
    long long row = (long long)blockIdx.x * 8 + wave;
    if (row >= nrows) return;
    float4 v = ((const float4*)(IN + row * H))[lane];
    float x0 = v.x > 0.0f ? v.x : (__expf(v.x) - 1.0f);
    float x1 = v.y > 0.0f ? v.y : (__expf(v.y) - 1.0f);
    float x2 = v.z > 0.0f ? v.z : (__expf(v.z) - 1.0f);
    float x3 = v.w > 0.0f ? v.w : (__expf(v.w) - 1.0f);
    float s = x0 + x1 + x2 + x3;
    float q = x0 * x0 + x1 * x1 + x2 * x2 + x3 * x3;
    #pragma unroll
    for (int mk = 16; mk >= 1; mk >>= 1) {
        s += __shfl_xor(s, mk, 32);
        q += __shfl_xor(q, mk, 32);
    }
    float mu  = s * (1.0f / 128.0f);
    float var = q * (1.0f / 128.0f) - mu * mu;
    float rs  = rsqrtf(var + 1e-5f);
    int c0 = lane * 4;
    float y0 = (x0 - mu) * rs * g[c0 + 0] + b[c0 + 0];
    float y1 = (x1 - mu) * rs * g[c0 + 1] + b[c0 + 1];
    float y2 = (x2 - mu) * rs * g[c0 + 2] + b[c0 + 2];
    float y3 = (x3 - mu) * rs * g[c0 + 3] + b[c0 + 3];
    float4 o = {y0, y1, y2, y3};
    ((float4*)(XA + row * H))[lane] = o;
    uint2 pk;
    pk.x = (unsigned)f2bf(y0) | ((unsigned)f2bf(y1) << 16);
    pk.y = (unsigned)f2bf(y2) | ((unsigned)f2bf(y3) << 16);
    *((uint2*)(Xb + row * H) + lane) = pk;
}

__global__ void pool_scatter(const float* __restrict__ XA, const int* __restrict__ batch,
                             float* __restrict__ pooled, float* __restrict__ cnt, int nrows) {
    int wave = threadIdx.x >> 5, lane = threadIdx.x & 31;
    long long row = (long long)blockIdx.x * 8 + wave;
    if (row >= nrows) return;
    int bsel = batch[row];
    float4 v = ((const float4*)(XA + row * H))[lane];
    int c0 = lane * 4;
    atomicAdd(&pooled[(size_t)bsel * H + c0 + 0], v.x);
    atomicAdd(&pooled[(size_t)bsel * H + c0 + 1], v.y);
    atomicAdd(&pooled[(size_t)bsel * H + c0 + 2], v.z);
    atomicAdd(&pooled[(size_t)bsel * H + c0 + 3], v.w);
    if (lane == 0) atomicAdd(&cnt[bsel], 1.0f);
}

__global__ void final_lin(const float* __restrict__ pooled, const float* __restrict__ cnt,
                          const float* __restrict__ Wl, const float* __restrict__ bl,
                          float* __restrict__ out) {
    int wave = threadIdx.x >> 5, lane = threadIdx.x & 31;
    int bsel = blockIdx.x * 8 + wave;
    if (bsel >= NB) return;
    float4 v = ((const float4*)(pooled + (size_t)bsel * H))[lane];
    int c0 = lane * 4;
    float s = v.x * Wl[c0] + v.y * Wl[c0 + 1] + v.z * Wl[c0 + 2] + v.w * Wl[c0 + 3];
    #pragma unroll
    for (int mk = 16; mk >= 1; mk >>= 1) s += __shfl_xor(s, mk, 32);
    if (lane == 0) out[bsel] = s / fmaxf(cnt[bsel], 1.0f) + bl[0];
}

// ---------------------------------------------------------------------------
extern "C" void kernel_launch(void* const* d_in, const int* in_sizes, int n_in,
                              void* d_out, int out_size, void* d_ws, size_t ws_size,
                              hipStream_t stream) {
    (void)n_in; (void)out_size; (void)ws_size;
    static const int TSZ[NTYPE]  = {100000, 20000, 120000, 60000, 100000, 50000, 50000};
    static const int TFEAT[NTYPE]= {4, 2, 10, 4, 8, 1, 1};
    static const int TOFF[NTYPE] = {0, 100000, 120000, 240000, 300000, 400000, 450000};
    static const int SRC_OFF[NREL] = {100000,240000,120000,120000,0,300000,300000,400000,450000,400000,450000,100000,120000,240000};
    static const int DST_OFF[NREL] = {0,0,0,300000,0,0,240000,300000,300000,400000,450000,100000,120000,240000};

    const int* esrc[NREL]; const int* edst[NREL]; int E[NREL];
    for (int r = 0; r < NREL; ++r) {
        const int* p = (const int*)d_in[7 + r];
        E[r] = in_sizes[7 + r] / 2;
        esrc[r] = p; edst[r] = p + E[r];
    }
    const int*   batch   = (const int*)d_in[21];
    const float* c1_root = (const float*)d_in[36];
    const float* c1_rel  = (const float*)d_in[37];
    const float* c1_bias = (const float*)d_in[38];
    const float* c2_root = (const float*)d_in[39];
    const float* c2_rel  = (const float*)d_in[40];
    const float* c2_bias = (const float*)d_in[41];
    const float* n1g = (const float*)d_in[42];
    const float* n1b = (const float*)d_in[43];
    const float* n2g = (const float*)d_in[44];
    const float* n2b = (const float*)d_in[45];
    const float* Wl  = (const float*)d_in[46];
    const float* bl  = (const float*)d_in[47];
    float* out = (float*)d_out;

    // workspace layout
    char* ws = (char*)d_ws;
    size_t off = 0;
    auto take = [&](size_t bytes) { size_t o = off; off = (off + bytes + 255) & ~(size_t)255; return o; };
    unsigned short* Xb   = (unsigned short*)(ws + take((size_t)NTOT * H * 2));
    float*          OUT  = (float*)(ws + take((size_t)(NTOT + 16) * H * 4));  // +dummy rows
    float*          XA   = (float*)(ws + take((size_t)NTOT * H * 4));
    float*          INVC = (float*)(ws + take((size_t)NREL * NTOT * 4));
    unsigned short* Wpk  = (unsigned short*)(ws + take((size_t)30 * 16384 * 2));
    float*          pooled = (float*)(ws + take((size_t)(NB * H + NB) * 4));
    float*          cntB = pooled + (size_t)NB * H;

    // 1) pack weights to bf16 B-fragment order
    pack_weights<<<(30 * 16384 + 255) / 256, 256, 0, stream>>>(c1_root, c1_rel, c2_root, c2_rel, Wpk);

    // 2) per-relation inverse mean counts (layer-invariant)
    long long ninv = (long long)NREL * NTOT;
    zero_f32<<<(unsigned)((ninv + 255) / 256), 256, 0, stream>>>(INVC, ninv);
    for (int r = 0; r < NREL; ++r)
        count_edges<<<(E[r] + 255) / 256, 256, 0, stream>>>(edst[r], E[r], DST_OFF[r], INVC + (size_t)r * NTOT);
    invert_counts<<<(unsigned)((ninv + 255) / 256), 256, 0, stream>>>(INVC, ninv);

    // 3) input projections -> bf16 features
    for (int t = 0; t < NTYPE; ++t) {
        const float* x  = (const float*)d_in[t];
        const float* Wt = (const float*)d_in[22 + 2 * t];
        const float* bt = (const float*)d_in[23 + 2 * t];
        long long nth = (long long)TSZ[t] * H;
        proj_type<<<(unsigned)((nth + 255) / 256), 256, 0, stream>>>(x, Wt, bt, Xb, TSZ[t], TFEAT[t], TOFF[t]);
    }

    // 4) layer 1: root GEMM + 14 edge-message GEMMs + ELU/LN
    gemm_root<<<NTOT / 32, 32, 0, stream>>>(Xb, Wpk, c1_bias, OUT);
    for (int r = 0; r < NREL; ++r)
        gemm_edges<<<(E[r] + 31) / 32, 32, 0, stream>>>(Xb, esrc[r], edst[r], E[r], SRC_OFF[r], DST_OFF[r],
                                                        Wpk + (size_t)(1 + r) * 16384,
                                                        INVC + (size_t)r * NTOT, OUT);
    act_ln<<<(NTOT + 7) / 8, 256, 0, stream>>>(OUT, n1g, n1b, XA, Xb, NTOT);

    // 5) layer 2
    gemm_root<<<NTOT / 32, 32, 0, stream>>>(Xb, Wpk + (size_t)15 * 16384, c2_bias, OUT);
    for (int r = 0; r < NREL; ++r)
        gemm_edges<<<(E[r] + 31) / 32, 32, 0, stream>>>(Xb, esrc[r], edst[r], E[r], SRC_OFF[r], DST_OFF[r],
                                                        Wpk + (size_t)(16 + r) * 16384,
                                                        INVC + (size_t)r * NTOT, OUT);
    act_ln<<<(NTOT + 7) / 8, 256, 0, stream>>>(OUT, n2g, n2b, XA, Xb, NTOT);

    // 6) segment-mean pooling over operator rows + final linear
    long long npool = (long long)NB * H + NB;
    zero_f32<<<(unsigned)((npool + 255) / 256), 256, 0, stream>>>(pooled, npool);
    pool_scatter<<<(100000 + 7) / 8, 256, 0, stream>>>(XA, batch, pooled, cntB, 100000);
    final_lin<<<(NB + 7) / 8, 256, 0, stream>>>(pooled, cntB, Wl, bl, out);
}
